// AttnBlock_15367392985367
// MI455X (gfx1250) — matrix operand order repaired and checked by reading, not compiled, not run on hardware
//
#include <hip/hip_runtime.h>
#include <hip/hip_bf16.h>

typedef __attribute__((ext_vector_type(16))) _Float16 v16h;
typedef __attribute__((ext_vector_type(8)))  _Float16 v8h;
typedef __attribute__((ext_vector_type(8)))  float    v8f;

#define WMMA_F16(A, B, C) \
  __builtin_amdgcn_wmma_f32_16x16x32_f16(false, (A), false, (B), (short)0, (C), false, false)

// Problem constants
constexpr int Cc = 128;       // channels
constexpr int Ss = 4096;      // spatial (16*16*16)
constexpr int Hh = 4;         // heads
constexpr int Dd = 32;        // dim per head
constexpr int Bb = 2;         // batch
constexpr float SCALE = 0.17677669529663688f; // 1/sqrt(32)

// ---------------------------------------------------------------------------
// Kernel 1: GroupNorm statistics. 64 (b,group) pairs, each over 4*4096 elems.
// stats[0..63] = mean, stats[64..127] = rstd
// ---------------------------------------------------------------------------
__global__ void gn_stats_kernel(const float* __restrict__ x, float* __restrict__ stats) {
  __shared__ float sh_s[256];
  __shared__ float sh_q[256];
  const int g = blockIdx.x;                      // b*32 + group
  const float* base = x + (size_t)g * 16384;     // 4 channels * 4096 spatial
  float s = 0.f, q = 0.f;
  for (int i = threadIdx.x; i < 16384; i += 256) {
    float t = base[i];
    s += t; q += t * t;
  }
  sh_s[threadIdx.x] = s; sh_q[threadIdx.x] = q;
  __syncthreads();
  for (int off = 128; off > 0; off >>= 1) {
    if ((int)threadIdx.x < off) {
      sh_s[threadIdx.x] += sh_s[threadIdx.x + off];
      sh_q[threadIdx.x] += sh_q[threadIdx.x + off];
    }
    __syncthreads();
  }
  if (threadIdx.x == 0) {
    float mean = sh_s[0] * (1.f / 16384.f);
    float var  = sh_q[0] * (1.f / 16384.f) - mean * mean;
    stats[g]      = mean;
    stats[64 + g] = rsqrtf(var + 1e-5f);
  }
}

// ---------------------------------------------------------------------------
// Kernel 2: fused GroupNorm-apply + QKV projection via WMMA.
// A = w_qkv tile (16 out x 32 in, f16), B = normalized x (32 in x 16 pos, f16)
// Q scaled by 1/sqrt(d). V stored transposed [bh][d][s] for the PV B-operand.
// ---------------------------------------------------------------------------
__global__ __launch_bounds__(128) void qkv_kernel(
    const float* __restrict__ x, const float* __restrict__ stats,
    const float* __restrict__ gamma, const float* __restrict__ beta,
    const float* __restrict__ wqkv,
    _Float16* __restrict__ qb, _Float16* __restrict__ kb, _Float16* __restrict__ vtb) {
  const int lane = threadIdx.x & 31;
  const int wave = threadIdx.x >> 5;
  const int hi   = lane >> 4;
  const int lo   = lane & 15;
  const int sBase = (blockIdx.x * 4 + wave) * 16;
  const int oBase = blockIdx.y * 16;             // 0..368
  const int b     = blockIdx.z;
  const float* mean = stats;
  const float* rstd = stats + 64;
  const int sPos = sBase + lo;

  v8f acc = {};
  #pragma unroll
  for (int kc = 0; kc < 128; kc += 32) {
    // A operand: 16x32 f16, contiguous k = 16*hi + s (lane lo = row, hi selects K chunks)
    const float* wrow = wqkv + (size_t)(oBase + lo) * Cc + kc;
    v16h a;
    #pragma unroll
    for (int i = 0; i < 8; ++i) {
      a[i]     = (_Float16)wrow[hi * 16 + i];
      a[8 + i] = (_Float16)wrow[hi * 16 + 8 + i];
    }
    // B operand: 32x16 f16, lane holds 16 consecutive channels at fixed pos
    v16h bm;
    #pragma unroll
    for (int j = 0; j < 16; ++j) {
      int c = kc + hi * 16 + j;
      float xv = x[((size_t)b * Cc + c) * Ss + sPos];
      int g = b * 32 + (c >> 2);
      float xn = (xv - mean[g]) * rstd[g] * gamma[c] + beta[c];
      bm[j] = (_Float16)xn;
    }
    acc = WMMA_F16(a, bm, acc);
  }
  // D layout: element r -> (M = hi*8+r, N = lo)
  #pragma unroll
  for (int r = 0; r < 8; ++r) {
    int o   = oBase + hi * 8 + r;
    int mat = o >> 7;                            // 0=q 1=k 2=v
    int oc  = o & 127;
    int h = oc >> 5, dd = oc & 31;
    float vv = acc[r];
    if (mat == 0) {
      qb[(((size_t)b * Hh + h) * Ss + sPos) * Dd + dd] = (_Float16)(vv * SCALE);
    } else if (mat == 1) {
      kb[(((size_t)b * Hh + h) * Ss + sPos) * Dd + dd] = (_Float16)vv;
    } else {
      vtb[(((size_t)b * Hh + h) * Dd + dd) * Ss + sPos] = (_Float16)vv;
    }
  }
}

// ---------------------------------------------------------------------------
// Kernel 3: flash attention. 8 waves/block; the block cooperatively stages
// each 32-key K tile and V^T tile into LDS with GLOBAL_LOAD_ASYNC_TO_LDS_B128
// (ASYNCcnt path), then every wave runs 2 QK^T WMMAs + online softmax +
// 2 PV WMMAs against the shared tiles.
// ---------------------------------------------------------------------------
__global__ __launch_bounds__(256) void flash_kernel(
    const _Float16* __restrict__ qb, const _Float16* __restrict__ kb,
    const _Float16* __restrict__ vtb, _Float16* __restrict__ ab) {
  __shared__ __align__(64) _Float16 sK[32][32];      // [key][d]     2 KB
  __shared__ __align__(64) _Float16 sVt[32][32];     // [d][key]     2 KB
  __shared__ __align__(32) float    sS[8][16][32];   // score tile per wave
  __shared__ __align__(32) _Float16 sP[8][16][32];   // probability tile per wave
  __shared__ float sAlpha[8][16];
  __shared__ float sL[8][16];

  const int tid  = threadIdx.x;
  const int lane = tid & 31;
  const int wave = tid >> 5;
  const int hi   = lane >> 4;
  const int lo   = lane & 15;
  const int bh   = blockIdx.y;                       // 0..7
  const int qBase = (blockIdx.x * 8 + wave) * 16;

  const _Float16* qbh = qb  + (size_t)bh * Ss * Dd;
  const _Float16* kbh = kb  + (size_t)bh * Ss * Dd;
  const _Float16* vbh = vtb + (size_t)bh * Dd * Ss;  // transposed [d][s]
  _Float16*       abh = ab  + (size_t)bh * Ss * Dd;

  // Q tile as A operand (loaded once, scale already folded in)
  const _Float16* qrow = qbh + (size_t)(qBase + lo) * Dd;
  v8h qlo8 = *(const v8h*)(qrow + hi * 16);
  v8h qhi8 = *(const v8h*)(qrow + hi * 16 + 8);
  v16h aq = __builtin_shufflevector(qlo8, qhi8, 0,1,2,3,4,5,6,7,8,9,10,11,12,13,14,15);

  // Per-thread staging assignment: threads 0-127 move the K tile (2 KB),
  // threads 128-255 move the V^T tile (2 KB); 16 bytes (B128) each.
  const unsigned ldsDst = (tid < 128)
      ? (unsigned)(uintptr_t)((char*)&sK[0][0]  + (size_t)tid * 16)
      : (unsigned)(uintptr_t)((char*)&sVt[0][0] + (size_t)(tid - 128) * 16);
  const int t2 = tid - 128;

  v8f o0 = {}, o1 = {};                              // out accum, d cols 0-15 / 16-31
  float m_state = -3.0e38f, l_state = 0.f;

  for (int kt = 0; kt < 128; ++kt) {
    const int kBase = kt * 32;

    // --- async-stage K / V^T tiles into LDS (one B128 per thread) ---
    const char* gsrc = (tid < 128)
        ? (const char*)kbh + (size_t)kBase * 64 + (size_t)tid * 16
        : (const char*)vbh + (size_t)(t2 >> 2) * 8192 + (size_t)(t2 & 3) * 16
                           + (size_t)kBase * 2;
    asm volatile("global_load_async_to_lds_b128 %0, %1, off"
                 :: "v"(ldsDst), "v"(gsrc) : "memory");
    if (kt + 1 < 128) {
      __builtin_prefetch(gsrc + ((tid < 128) ? 2048 : 64), 0, 0);
    }
    asm volatile("s_wait_asynccnt 0x0" ::: "memory");
    __syncthreads();                                 // tiles visible to all waves

    // --- QK^T: B operands straight from shared K tile ---
    v16h bk0 = *(const v16h*)&sK[lo][hi * 16];
    v16h bk1 = *(const v16h*)&sK[16 + lo][hi * 16];
    v8f s0 = {}, s1 = {};
    s0 = WMMA_F16(aq, bk0, s0);
    s1 = WMMA_F16(aq, bk1, s1);

    // scatter scores to LDS: (M = hi*8+r, N = lo)
    #pragma unroll
    for (int r = 0; r < 8; ++r) {
      sS[wave][hi * 8 + r][lo]      = s0[r];
      sS[wave][hi * 8 + r][16 + lo] = s1[r];
    }
    __syncthreads();

    // per-row online softmax: lanes 0-15 each own one query row
    if (lane < 16) {
      const float* rowp = &sS[wave][lane][0];
      float mloc = -3.0e38f;
      #pragma unroll
      for (int j = 0; j < 32; ++j) mloc = fmaxf(mloc, rowp[j]);
      float mnew  = fmaxf(m_state, mloc);
      float alpha = __expf(m_state - mnew);
      float lsum = 0.f;
      #pragma unroll
      for (int j = 0; j < 32; ++j) {
        float p = __expf(rowp[j] - mnew);
        lsum += p;
        sP[wave][lane][j] = (_Float16)p;
      }
      l_state = l_state * alpha + lsum;
      m_state = mnew;
      sAlpha[wave][lane] = alpha;
    }
    __syncthreads();

    // rescale accumulators by per-row alpha
    #pragma unroll
    for (int r = 0; r < 8; ++r) {
      float al = sAlpha[wave][hi * 8 + r];
      o0[r] *= al;
      o1[r] *= al;
    }
    // P as A operand from LDS
    v8h plo = *(const v8h*)&sP[wave][lo][hi * 16];
    v8h phi = *(const v8h*)&sP[wave][lo][hi * 16 + 8];
    v16h ap = __builtin_shufflevector(plo, phi, 0,1,2,3,4,5,6,7,8,9,10,11,12,13,14,15);
    // V^T as B operands from shared tile (contiguous along key index)
    v16h bv0 = *(const v16h*)&sVt[lo][hi * 16];
    v16h bv1 = *(const v16h*)&sVt[16 + lo][hi * 16];
    o0 = WMMA_F16(ap, bv0, o0);
    o1 = WMMA_F16(ap, bv1, o1);
    __syncthreads();                                 // tiles free for next stage
  }

  if (lane < 16) sL[wave][lane] = l_state;
  __syncthreads();
  #pragma unroll
  for (int r = 0; r < 8; ++r) {
    float linv = 1.f / sL[wave][hi * 8 + r];
    int s = qBase + hi * 8 + r;
    _Float16* orow = abh + (size_t)s * Dd;
    orow[lo]      = (_Float16)(o0[r] * linv);
    orow[16 + lo] = (_Float16)(o1[r] * linv);
  }
}

// ---------------------------------------------------------------------------
// Kernel 4: output projection y = w_out @ attn + b_out via WMMA.
// ---------------------------------------------------------------------------
__global__ __launch_bounds__(128) void proj_kernel(
    const _Float16* __restrict__ ab, const float* __restrict__ wout,
    const float* __restrict__ bout, float* __restrict__ y) {
  const int lane = threadIdx.x & 31;
  const int wave = threadIdx.x >> 5;
  const int hi   = lane >> 4;
  const int lo   = lane & 15;
  const int sBase = (blockIdx.x * 4 + wave) * 16;
  const int oBase = blockIdx.y * 16;
  const int b     = blockIdx.z;
  const int sPos = sBase + lo;

  v8f acc = {};
  #pragma unroll
  for (int kc = 0; kc < 128; kc += 32) {
    const float* wrow = wout + (size_t)(oBase + lo) * Cc + kc;
    v16h a;
    #pragma unroll
    for (int i = 0; i < 8; ++i) {
      a[i]     = (_Float16)wrow[hi * 16 + i];
      a[8 + i] = (_Float16)wrow[hi * 16 + 8 + i];
    }
    int cBase = kc + hi * 16;                          // 16 chans within one head
    const _Float16* bp =
        ab + (((size_t)b * Hh + (cBase >> 5)) * Ss + sPos) * Dd + (cBase & 31);
    v16h bm = *(const v16h*)bp;
    acc = WMMA_F16(a, bm, acc);
  }
  #pragma unroll
  for (int r = 0; r < 8; ++r) {
    int o = oBase + hi * 8 + r;
    y[((size_t)b * Cc + o) * Ss + sPos] = acc[r] + bout[o];
  }
}

// ---------------------------------------------------------------------------
extern "C" void kernel_launch(void* const* d_in, const int* in_sizes, int n_in,
                              void* d_out, int out_size, void* d_ws, size_t ws_size,
                              hipStream_t stream) {
  const float* x     = (const float*)d_in[0];
  const float* gamma = (const float*)d_in[1];
  const float* beta  = (const float*)d_in[2];
  const float* wqkv  = (const float*)d_in[3];
  const float* wout  = (const float*)d_in[4];
  const float* bout  = (const float*)d_in[5];
  float* y = (float*)d_out;

  const size_t headElems = (size_t)Bb * Hh * Ss * Dd;  // 1,048,576
  float*    stats = (float*)d_ws;
  _Float16* qb  = (_Float16*)((char*)d_ws + 1024);
  _Float16* kb  = qb  + headElems;
  _Float16* vtb = kb  + headElems;
  _Float16* ab  = vtb + headElems;

  gn_stats_kernel<<<64, 256, 0, stream>>>(x, stats);
  qkv_kernel<<<dim3(64, 24, 2), 128, 0, stream>>>(x, stats, gamma, beta, wqkv, qb, kb, vtb);
  flash_kernel<<<dim3(32, 8), 256, 0, stream>>>(qb, kb, vtb, ab);
  proj_kernel<<<dim3(64, 8, 2), 128, 0, stream>>>(ab, wout, bout, y);
}
